// GraphConvClassifier_71184787964265
// MI455X (gfx1250) — matrix operand order repaired and checked by reading, not compiled, not run on hardware
//
#include <hip/hip_runtime.h>
#include <hip/hip_bf16.h>
#include <stdint.h>

#define NN 50000      // nodes
#define NE 800000     // edges
#define HD 128        // in/hid dim
#define NG 64         // graphs
#define NC 2          // classes
#define EPSI 1e-5f

typedef __attribute__((ext_vector_type(16))) __bf16 v16bf;
typedef __attribute__((ext_vector_type(8)))  float  v8f;

// ---------- helpers ----------
__device__ __forceinline__ void atomAddF(float* p, float v) {
    unsafeAtomicAdd(p, v);   // hardware global_atomic_add_f32 (no CAS loop)
}

// ---------- generic fill ----------
__global__ void fill_f32(float* __restrict__ p, float v, int n) {
    int t = blockIdx.x * blockDim.x + threadIdx.x;
    if (t < n) p[t] = v;
}

// ---------- pack W (fp32 row-major [K=128][N=128]) into WMMA B-fragment bf16 hi/lo ----------
// frag index: (((ks*8 + nt)*32 + lane)*16 + e); k = ks*32 + (lane<16 ? 0 : 8) + (e<8 ? e : 8+e); n = nt*16 + (lane&15)
__global__ void pack_w(const float* __restrict__ W1, const float* __restrict__ W2,
                       unsigned short* __restrict__ hi1, unsigned short* __restrict__ lo1,
                       unsigned short* __restrict__ hi2, unsigned short* __restrict__ lo2) {
    int t = blockIdx.x * blockDim.x + threadIdx.x;
    if (t >= 2 * 16384) return;
    const float* W = (t < 16384) ? W1 : W2;
    unsigned short* hi = (t < 16384) ? hi1 : hi2;
    unsigned short* lo = (t < 16384) ? lo1 : lo2;
    int idx  = t & 16383;
    int e    = idx & 15;
    int lane = (idx >> 4) & 31;
    int nt   = (idx >> 9) & 7;
    int ks   = idx >> 12;
    int k = ks * 32 + ((lane & 16) ? 8 : 0) + (e < 8 ? e : e + 8);
    int n = nt * 16 + (lane & 15);
    float v = W[k * HD + n];
    __bf16 h = (__bf16)v;                 // RNE hardware convert
    __bf16 l = (__bf16)(v - (float)h);
    hi[idx] = __builtin_bit_cast(unsigned short, h);
    lo[idx] = __builtin_bit_cast(unsigned short, l);
}

// ---------- WMMA GEMM: C[N x 128] = A[N x 128] @ W, bf16 hi/lo split, f32 accum ----------
// One wave computes a 16(M) x 64(N) strip: task = (M-tile, column-half).
__global__ void __launch_bounds__(256) gemm_wmma(const float* __restrict__ A,
                                                 const unsigned short* __restrict__ Bhi,
                                                 const unsigned short* __restrict__ Blo,
                                                 float* __restrict__ C, int nTiles) {
    int lane = threadIdx.x & 31;
    int wid  = threadIdx.x >> 5;
    int upper = (lane & 16) ? 1 : 0;
    int nTasks = nTiles * 2;

    for (int task = blockIdx.x * 8 + wid; task < nTasks; task += gridDim.x * 8) {
        int tile = task >> 1;
        int nh   = task & 1;          // column half: n-tiles nh*4 .. nh*4+3

        v8f acc[4];
#pragma unroll
        for (int j = 0; j < 4; ++j) acc[j] = (v8f){0.f,0.f,0.f,0.f,0.f,0.f,0.f,0.f};

        const float* arow = A + (size_t)(tile * 16 + (lane & 15)) * HD;

#pragma unroll
        for (int ks = 0; ks < 4; ++ks) {
            // A-fragment: 16-bit A 16x32 layout — lane<16 holds K {0..7,16..23}, lane>=16 holds {8..15,24..31}
            int k0 = ks * 32 + (upper ? 8 : 0);
            float cv[16];
            float4 p;
            p = *(const float4*)(arow + k0);       cv[0]=p.x; cv[1]=p.y; cv[2]=p.z; cv[3]=p.w;
            p = *(const float4*)(arow + k0 + 4);   cv[4]=p.x; cv[5]=p.y; cv[6]=p.z; cv[7]=p.w;
            p = *(const float4*)(arow + k0 + 16);  cv[8]=p.x; cv[9]=p.y; cv[10]=p.z; cv[11]=p.w;
            p = *(const float4*)(arow + k0 + 20);  cv[12]=p.x; cv[13]=p.y; cv[14]=p.z; cv[15]=p.w;
            v16bf ahi, alo;
#pragma unroll
            for (int e = 0; e < 16; ++e) {
                __bf16 h = (__bf16)cv[e];
                ahi[e] = h;
                alo[e] = (__bf16)(cv[e] - (float)h);
            }
#pragma unroll
            for (int j = 0; j < 4; ++j) {
                int nt = nh * 4 + j;
                size_t boff = (size_t)(((ks * 8 + nt) * 32 + lane)) * 16;
                v16bf bhi = *(const v16bf*)(Bhi + boff);
                v16bf blo = *(const v16bf*)(Blo + boff);
                acc[j] = __builtin_amdgcn_wmma_f32_16x16x32_bf16(false, ahi, false, bhi,
                                                                 (short)0, acc[j], false, false);
                acc[j] = __builtin_amdgcn_wmma_f32_16x16x32_bf16(false, alo, false, bhi,
                                                                 (short)0, acc[j], false, false);
                acc[j] = __builtin_amdgcn_wmma_f32_16x16x32_bf16(false, ahi, false, blo,
                                                                 (short)0, acc[j], false, false);
            }
        }
        // C layout: VGPR r -> M = r + 8*upper, N = nt*16 + (lane&15)
        int rbase = tile * 16 + (upper ? 8 : 0);
        int col0  = nh * 64 + (lane & 15);
#pragma unroll
        for (int j = 0; j < 4; ++j)
#pragma unroll
            for (int r = 0; r < 8; ++r)
                C[(size_t)(rbase + r) * HD + col0 + j * 16] = acc[j][r];
    }
}

// ---------- degree / counts ----------
__global__ void deg_edges(const int* __restrict__ dst, float* __restrict__ deg, int nE) {
    int t = blockIdx.x * blockDim.x + threadIdx.x;
    if (t < nE) atomAddF(&deg[dst[t]], 1.0f);
}
__global__ void counts_nodes(const int* __restrict__ batch, float* __restrict__ counts, int n) {
    int t = blockIdx.x * blockDim.x + threadIdx.x;
    if (t < n) atomAddF(&counts[batch[t]], 1.0f);
}
__global__ void finalize_dinv(float* __restrict__ deg, int n) {
    int t = blockIdx.x * blockDim.x + threadIdx.x;
    if (t < n) deg[t] = rsqrtf(fmaxf(deg[t], 1e-12f));
}

// ---------- self-loop term + bias: Out[i][c] = b[c] + H[i][c]*dinv[i]^2 ----------
__global__ void self_init(const float* __restrict__ H, const float* __restrict__ dinv,
                          const float* __restrict__ b, float* __restrict__ Out, int n) {
    size_t t = (size_t)blockIdx.x * blockDim.x + threadIdx.x;
    if (t >= (size_t)n * HD) return;
    int i = (int)(t >> 7), c = (int)(t & 127);
    float di = dinv[i];
    Out[t] = b[c] + H[t] * di * di;
}

// ---------- edge scatter: Out[dst] += H[src]*dinv[src]*dinv[dst]; one wave per edge ----------
__global__ void __launch_bounds__(256) edge_agg(const float* __restrict__ H,
                                                const int* __restrict__ src, const int* __restrict__ dst,
                                                const float* __restrict__ dinv,
                                                float* __restrict__ Out, int nE) {
    int t = blockIdx.x * 256 + threadIdx.x;
    int e = t >> 5;
    if (e >= nE) return;
    int lane = t & 31;
    int s = src[e], d = dst[e];
    float w = dinv[s] * dinv[d];
    float4 v = *(const float4*)(H + (size_t)s * HD + lane * 4);
    float* o = Out + (size_t)d * HD + lane * 4;
    atomAddF(o + 0, v.x * w);
    atomAddF(o + 1, v.y * w);
    atomAddF(o + 2, v.z * w);
    atomAddF(o + 3, v.w * w);
}

// ---------- per-feature sum / sumsq (InstanceNorm stats) ----------
__global__ void __launch_bounds__(256) col_stats(const float* __restrict__ H,
                                                 float* __restrict__ stats, int nNodes) {
    __shared__ float ls[256], lq[256];
    int c = threadIdx.x & 127;
    int half = threadIdx.x >> 7;
    float s = 0.f, q = 0.f;
    for (int i = blockIdx.x * 2 + half; i < nNodes; i += gridDim.x * 2) {
        float v = H[(size_t)i * HD + c];
        s += v; q += v * v;
    }
    ls[threadIdx.x] = s; lq[threadIdx.x] = q;
    __syncthreads();
    if (threadIdx.x < 128) {
        atomAddF(&stats[c],       ls[threadIdx.x] + ls[threadIdx.x + 128]);
        atomAddF(&stats[128 + c], lq[threadIdx.x] + lq[threadIdx.x + 128]);
    }
}

// ---------- norm + relu ----------
__global__ void norm_relu(const float* __restrict__ In, const float* __restrict__ stats,
                          float* __restrict__ Out, int nNodes) {
    size_t t = (size_t)blockIdx.x * blockDim.x + threadIdx.x;
    if (t >= (size_t)nNodes * HD) return;
    int c = (int)(t & 127);
    float inv_n = 1.0f / (float)nNodes;
    float mu  = stats[c] * inv_n;
    float var = stats[128 + c] * inv_n - mu * mu;
    float v = (In[t] - mu) * rsqrtf(var + EPSI);
    Out[t] = v > 0.f ? v : 0.f;
}

// ---------- norm + relu fused with mean-pool accumulation ----------
__global__ void norm_relu_pool(const float* __restrict__ In, const float* __restrict__ stats,
                               const int* __restrict__ batch, float* __restrict__ pooled, int nNodes) {
    size_t t = (size_t)blockIdx.x * blockDim.x + threadIdx.x;
    if (t >= (size_t)nNodes * HD) return;
    int i = (int)(t >> 7), c = (int)(t & 127);
    float inv_n = 1.0f / (float)nNodes;
    float mu  = stats[c] * inv_n;
    float var = stats[128 + c] * inv_n - mu * mu;
    float v = (In[t] - mu) * rsqrtf(var + EPSI);
    v = v > 0.f ? v : 0.f;
    atomAddF(&pooled[(size_t)batch[i] * HD + c], v);
}

// ---------- final FC: out[g][j] = fcb[j] + sum_c pooled[g][c]/cnt * fcW[c][j] ----------
__global__ void fc_kernel(const float* __restrict__ pooled, const float* __restrict__ counts,
                          const float* __restrict__ fcW, const float* __restrict__ fcb,
                          float* __restrict__ out) {
    int t = threadIdx.x;
    if (t >= NG * NC) return;
    int g = t >> 1, j = t & 1;
    float inv = 1.0f / fmaxf(counts[g], 1.0f);
    float acc = fcb[j];
    for (int c = 0; c < HD; ++c)
        acc += pooled[g * HD + c] * inv * fcW[c * NC + j];
    out[g * NC + j] = acc;
}

// ---------- host ----------
static inline size_t alignup(size_t x) { return (x + 255) & ~(size_t)255; }

extern "C" void kernel_launch(void* const* d_in, const int* in_sizes, int n_in,
                              void* d_out, int out_size, void* d_ws, size_t ws_size,
                              hipStream_t stream) {
    const float* x     = (const float*)d_in[0];
    const int*   eidx  = (const int*)d_in[1];
    const int*   batch = (const int*)d_in[2];
    const float* W1    = (const float*)d_in[3];
    const float* b1    = (const float*)d_in[4];
    const float* W2    = (const float*)d_in[5];
    const float* b2    = (const float*)d_in[6];
    const float* fcW   = (const float*)d_in[7];
    const float* fcb   = (const float*)d_in[8];
    float* out = (float*)d_out;
    const int* src = eidx;
    const int* dst = eidx + NE;

    char* w = (char*)d_ws;
    float* bufA   = (float*)w;          w += alignup((size_t)NN * HD * 4);
    float* bufB   = (float*)w;          w += alignup((size_t)NN * HD * 4);
    float* dinv   = (float*)w;          w += alignup((size_t)NN * 4);
    float* stats  = (float*)w;          w += alignup(256 * 4);
    float* pooled = (float*)w;          w += alignup((size_t)NG * HD * 4);
    float* counts = (float*)w;          w += alignup((size_t)NG * 4);
    unsigned short* hi1 = (unsigned short*)w; w += alignup(16384 * 2);
    unsigned short* lo1 = (unsigned short*)w; w += alignup(16384 * 2);
    unsigned short* hi2 = (unsigned short*)w; w += alignup(16384 * 2);
    unsigned short* lo2 = (unsigned short*)w; w += alignup(16384 * 2);

    const int TPB = 256;
    const int nTiles = NN / 16;                       // 3125
    const int nTasks = nTiles * 2;                    // 6250 wave-tasks
    dim3 gGemm((nTasks + 7) / 8), bTPB(TPB);
    size_t nhd = (size_t)NN * HD;
    int gNHD = (int)((nhd + TPB - 1) / TPB);

    // weight packing + degree/count setup
    pack_w<<<128, TPB, 0, stream>>>(W1, W2, hi1, lo1, hi2, lo2);
    fill_f32<<<(NN + TPB - 1) / TPB, TPB, 0, stream>>>(dinv, 1.0f, NN);   // self-loop degree
    fill_f32<<<1, TPB, 0, stream>>>(counts, 0.0f, NG);
    fill_f32<<<(NG * HD + TPB - 1) / TPB, TPB, 0, stream>>>(pooled, 0.0f, NG * HD);
    fill_f32<<<1, TPB, 0, stream>>>(stats, 0.0f, 256);
    deg_edges<<<(NE + TPB - 1) / TPB, TPB, 0, stream>>>(dst, dinv, NE);
    counts_nodes<<<(NN + TPB - 1) / TPB, TPB, 0, stream>>>(batch, counts, NN);
    finalize_dinv<<<(NN + TPB - 1) / TPB, TPB, 0, stream>>>(dinv, NN);

    // layer 1
    gemm_wmma<<<gGemm, bTPB, 0, stream>>>(x, hi1, lo1, bufA, nTiles);
    self_init<<<gNHD, TPB, 0, stream>>>(bufA, dinv, b1, bufB, NN);
    edge_agg<<<(NE * 32 + TPB - 1) / TPB, TPB, 0, stream>>>(bufA, src, dst, dinv, bufB, NE);
    col_stats<<<512, TPB, 0, stream>>>(bufB, stats, NN);
    norm_relu<<<gNHD, TPB, 0, stream>>>(bufB, stats, bufA, NN);

    // layer 2
    fill_f32<<<1, TPB, 0, stream>>>(stats, 0.0f, 256);
    gemm_wmma<<<gGemm, bTPB, 0, stream>>>(bufA, hi2, lo2, bufB, nTiles);
    self_init<<<gNHD, TPB, 0, stream>>>(bufB, dinv, b2, bufA, NN);
    edge_agg<<<(NE * 32 + TPB - 1) / TPB, TPB, 0, stream>>>(bufB, src, dst, dinv, bufA, NE);
    col_stats<<<512, TPB, 0, stream>>>(bufA, stats, NN);
    norm_relu_pool<<<gNHD, TPB, 0, stream>>>(bufA, stats, batch, pooled, NN);

    // readout
    fc_kernel<<<1, 128, 0, stream>>>(pooled, counts, fcW, fcb, out);
    (void)in_sizes; (void)n_in; (void)out_size; (void)ws_size;
}
